// LRSA_40381282517291
// MI455X (gfx1250) — compile-verified, matrix operand order checked by reading, not run on hardware
//
#include <hip/hip_runtime.h>
#include <hip/hip_bf16.h>
#include <math.h>
#include <stdint.h>

typedef _Float16 f16;
typedef __attribute__((ext_vector_type(16))) _Float16 v16h;
typedef __attribute__((ext_vector_type(8)))  _Float16 v8h;
typedef __attribute__((ext_vector_type(4)))  _Float16 v4h;
typedef __attribute__((ext_vector_type(8)))  float    v8f;

#define BDIM 4
#define CDIM 64
#define HDIM 256
#define WDIM 256
#define HWSZ (HDIM * WDIM)
#define PS   16
#define STEP 14
#define NPH  19
#define NPW  19
#define NTOK 256
#define C2   128
#define HALO 20

__device__ __forceinline__ float gelu_exact(float x) {
  return 0.5f * x * (1.0f + erff(x * 0.70710678118654752f));
}

// ---------------- CDNA5 async global->LDS helpers (gfx1250) -------------------
__device__ __forceinline__ void async_load_b128(uint32_t lds_byte_off, const void* g) {
  asm volatile("global_load_async_to_lds_b128 %0, %1, off"
               :: "v"(lds_byte_off), "v"((unsigned long long)(uintptr_t)g)
               : "memory");
}
__device__ __forceinline__ void async_load_b64(uint32_t lds_byte_off, const void* g) {
  asm volatile("global_load_async_to_lds_b64 %0, %1, off"
               :: "v"(lds_byte_off), "v"((unsigned long long)(uintptr_t)g)
               : "memory");
}
__device__ __forceinline__ void wait_async() {
  asm volatile("s_wait_asynccnt 0x0" ::: "memory");
}

// ---------------- CDNA5 WMMA f16 16x16x32 fragment helpers (wave32) -----------
// A matrix 16x32 row-major source: per-lane halves are two contiguous 16B runs
// (K 0..7 / 16..23 for lanes 0-15; +8 for lanes 16-31) -> ds_load_b128 pairs.
__device__ __forceinline__ v16h frag_a_rowmajor(const f16* buf, int stride,
                                                int mbase, int kbase, int lane) {
  const int row = mbase + (lane & 15);
  const int khi = (lane >> 4) << 3;
  v16h a;
#pragma unroll
  for (int h = 0; h < 16; ++h) {
    const int k = kbase + h + ((h >> 3) << 3) + khi;
    a[h] = buf[row * stride + k];
  }
  return a;
}
// Swizzled B layout: fragment for (ktile,ntile) stored as 512 contiguous halves,
// lane-major (lane*16 + h) -> each lane reads 32 contiguous bytes.
__device__ __forceinline__ v16h frag_b_swizzled(const f16* buf, int ntiles,
                                                int kbase, int nbase, int lane) {
  const f16* p = buf + ((((kbase >> 5) * ntiles) + (nbase >> 4)) << 9) + (lane << 4);
  v16h b;
#pragma unroll
  for (int h = 0; h < 16; ++h) b[h] = p[h];
  return b;
}
// Index of element B[k][n] inside the swizzled layout (inverse of frag load).
__device__ __forceinline__ int bsw_index(int k, int n, int ntiles) {
  return ((((k >> 5) * ntiles) + (n >> 4)) << 9) |
         ((((n & 15) | (((k >> 3) & 1) << 4))) << 4) |
         ((k & 7) | (((k >> 4) & 1) << 3));
}
// Stage a row-major f32 weight W[K][N] into the swizzled f16 B layout.
__device__ __forceinline__ void stage_weight_swizzled(f16* dst, const float* W,
                                                      int K, int N, int tid) {
  const int ntiles = N >> 4;
  for (int i = tid; i < K * N; i += 256) {
    const int tile = i >> 9, rem = i & 511;
    const int lane_ = rem >> 4, h = rem & 15;
    const int kt = tile / ntiles, nt = tile - kt * ntiles;
    const int col = (nt << 4) + (lane_ & 15);
    const int k = (kt << 5) + h + ((h >> 3) << 3) + ((lane_ >> 4) << 3);
    dst[i] = (f16)W[k * N + col];
  }
}
// One 16x16 output tile of [256x64] = A[256x64] @ Bsw[64x64]
__device__ __forceinline__ v8f gemm64_tile(const f16* A, const f16* Bsw,
                                           int mt, int nt, int lane) {
  v8f acc = {};
#pragma unroll
  for (int kk = 0; kk < CDIM; kk += 32) {
    v16h a  = frag_a_rowmajor(A, CDIM, mt, kk, lane);
    v16h bm = frag_b_swizzled(Bsw, 4, kk, nt, lane);
    acc = __builtin_amdgcn_wmma_f32_16x16x32_f16(false, a, false, bm,
                                                 (short)0, acc, false, false);
  }
  return acc;
}

// ---------------- kernel 0: zero overlap accumulator + build overlap mask -----
__global__ void lrsa_init_kernel(float* __restrict__ yacc, float* __restrict__ mask) {
  const int idx = blockIdx.x * 256 + threadIdx.x;
  yacc[idx] = 0.0f; // grid sized exactly B*C*H*W/256
  if (blockIdx.x == 0 && threadIdx.x < HDIM) {
    const int h = threadIdx.x;
    float m = 1.0f;
    for (int i = STEP; i < HDIM + STEP - PS; i += STEP) {
      int top = i;
      if (top + PS > HDIM) top = HDIM - PS;
      const int down = i + PS - STEP;
      if (h >= top && h < down) m *= 0.5f;
    }
    mask[h] = m;
  }
}

// ---------------- kernel 1: per-patch gated attention -------------------------
__global__ __launch_bounds__(256, 1)
void lrsa_attn_kernel(const float* __restrict__ x,
                      const float* __restrict__ ln1_w, const float* __restrict__ ln1_b,
                      const float* __restrict__ Wg, const float* __restrict__ bg,
                      const float* __restrict__ Wq, const float* __restrict__ Wk,
                      const float* __restrict__ Wv,
                      const float* __restrict__ pe_w, const float* __restrict__ pe_b,
                      const float* __restrict__ Wproj,
                      float* __restrict__ yacc) {
  extern __shared__ char smem[];
  f16* XN = (f16*)(smem);              // 256x64  f16 row-major (reused as OF)
  f16* QF = (f16*)(smem + 32768);      // 256x64  row-major (A + pe conv)
  f16* KF = (f16*)(smem + 65536);      // swizzled B for q@k^T (K=64,N=256)
  f16* VF = (f16*)(smem + 98304);      // swizzled B for P@V   (K=256,N=64)
  f16* ZF = (f16*)(smem + 131072);     // 256x64 row-major gate
  f16* SM = (f16*)(smem + 163840);     // scores 256x256 (first reused as RAW f32)
  f16* WB = (f16*)(smem + 294912);     // swizzled 64x64 weight

  const int tid  = threadIdx.x;
  const int lane = tid & 31;
  const int wave = tid >> 5;
  const uint32_t ldsbase = __builtin_amdgcn_groupstaticsize();

  const int wg = blockIdx.x;
  const int b  = wg / (NPH * NPW);
  const int pr = wg % (NPH * NPW);
  int top  = (pr / NPW) * STEP; if (top  > HDIM - PS) top  = HDIM - PS;
  int left = (pr % NPW) * STEP; if (left > WDIM - PS) left = WDIM - PS;

  __builtin_prefetch(Wg, 0, 0);
  __builtin_prefetch(Wq, 0, 0);
  __builtin_prefetch(Wk, 0, 0);
  __builtin_prefetch(Wv, 0, 0);
  __builtin_prefetch(Wproj, 0, 0);

  // ---- Phase A0: async-copy raw f32 patch into LDS (SM region), 8B chunks ----
  {
    const float* xb = x + (size_t)b * CDIM * HWSZ + (size_t)top * WDIM + left;
    const uint32_t rawoff = ldsbase + 163840u;
    for (int it = 0; it < 32; ++it) {
      const int f = tid + (it << 8);      // 8192 chunks of 8 bytes
      const int c = f >> 7;               // 128 chunks per channel
      const int row = (f >> 3) & 15;
      const int ch = f & 7;
      async_load_b64(rawoff + ((uint32_t)f << 3),
                     xb + (size_t)c * HWSZ + row * WDIM + (ch << 1));
    }
    wait_async();
  }
  __syncthreads();

  // ---- Phase A1: LayerNorm from LDS -> XN (f16) ----
  {
    const float* RAWf = (const float*)(smem + 163840);
    const int tok = tid;
    float v[CDIM];
    float mu = 0.0f;
#pragma unroll
    for (int c = 0; c < CDIM; ++c) { v[c] = RAWf[(c << 8) + tok]; mu += v[c]; }
    mu *= (1.0f / CDIM);
    float var = 0.0f;
#pragma unroll
    for (int c = 0; c < CDIM; ++c) { const float d = v[c] - mu; var += d * d; }
    var *= (1.0f / CDIM);
    const float rs = rsqrtf(var + 1e-6f);
#pragma unroll
    for (int c = 0; c < CDIM; ++c)
      XN[tok * CDIM + c] = (f16)((v[c] - mu) * rs * ln1_w[c] + ln1_b[c]);
  }
  __syncthreads();

  // ---- Phase B: four fully specialized GEMM passes (no runtime dispatch) ----
  // B1: z = gelu(xn @ Wg + bg) -> ZF row-major
  stage_weight_swizzled(WB, Wg, CDIM, CDIM, tid);
  __syncthreads();
  for (int t = wave; t < 64; t += 8) {
    const int mt = (t >> 2) << 4, nt = (t & 3) << 4;
    const v8f acc = gemm64_tile(XN, WB, mt, nt, lane);
    const int col = nt + (lane & 15);
    const int mb  = mt + ((lane >> 4) << 3);
    const float bias = bg[col];
#pragma unroll
    for (int r = 0; r < 8; ++r)
      ZF[(mb + r) * CDIM + col] = (f16)gelu_exact(acc[r] + bias);
  }
  __syncthreads();
  // B2: q = xn @ Wq -> QF row-major (A operand + pe conv source)
  stage_weight_swizzled(WB, Wq, CDIM, CDIM, tid);
  __syncthreads();
  for (int t = wave; t < 64; t += 8) {
    const int mt = (t >> 2) << 4, nt = (t & 3) << 4;
    const v8f acc = gemm64_tile(XN, WB, mt, nt, lane);
    const int col = nt + (lane & 15);
    const int mb  = mt + ((lane >> 4) << 3);
#pragma unroll
    for (int r = 0; r < 8; ++r)
      QF[(mb + r) * CDIM + col] = (f16)acc[r];
  }
  __syncthreads();
  // B3: k = xn @ Wk -> KF as swizzled B for q@k^T: B[k=chan][n=token], N=256
  stage_weight_swizzled(WB, Wk, CDIM, CDIM, tid);
  __syncthreads();
  for (int t = wave; t < 64; t += 8) {
    const int mt = (t >> 2) << 4, nt = (t & 3) << 4;
    const v8f acc = gemm64_tile(XN, WB, mt, nt, lane);
    const int col = nt + (lane & 15);
    const int mb  = mt + ((lane >> 4) << 3);
#pragma unroll
    for (int r = 0; r < 8; ++r)
      KF[bsw_index(col, mb + r, 16)] = (f16)acc[r];
  }
  __syncthreads();
  // B4: v = xn @ Wv -> VF as swizzled B for P@V: B[k=token][n=chan], N=64
  stage_weight_swizzled(WB, Wv, CDIM, CDIM, tid);
  __syncthreads();
  for (int t = wave; t < 64; t += 8) {
    const int mt = (t >> 2) << 4, nt = (t & 3) << 4;
    const v8f acc = gemm64_tile(XN, WB, mt, nt, lane);
    const int col = nt + (lane & 15);
    const int mb  = mt + ((lane >> 4) << 3);
#pragma unroll
    for (int r = 0; r < 8; ++r)
      VF[bsw_index(mb + r, col, 4)] = (f16)acc[r];
  }
  __syncthreads();

  // ---- Phase C: S = (q @ k^T) * C^-0.5 ----
  for (int t = wave; t < 256; t += 8) {
    const int mt = (t >> 4) << 4, nt = (t & 15) << 4;
    v8f acc = {};
#pragma unroll
    for (int kk = 0; kk < CDIM; kk += 32) {
      v16h a  = frag_a_rowmajor(QF, CDIM, mt, kk, lane);
      v16h bm = frag_b_swizzled(KF, 16, kk, nt, lane);
      acc = __builtin_amdgcn_wmma_f32_16x16x32_f16(false, a, false, bm,
                                                   (short)0, acc, false, false);
    }
    const int col = nt + (lane & 15);
    const int mb  = mt + ((lane >> 4) << 3);
#pragma unroll
    for (int r = 0; r < 8; ++r)
      SM[(mb + r) * NTOK + col] = (f16)(acc[r] * 0.125f);
  }
  __syncthreads();

  // ---- Phase D: row softmax, vectorized v8h (b128 LDS traffic) ----
  {
    v8h* rowv = (v8h*)(SM + tid * NTOK);
    float mx = -1e30f;
    for (int j = 0; j < 32; ++j) {
      const v8h c = rowv[j];
#pragma unroll
      for (int e = 0; e < 8; ++e) mx = fmaxf(mx, (float)c[e]);
    }
    float s = 0.0f;
    for (int j = 0; j < 32; ++j) {
      v8h c = rowv[j];
#pragma unroll
      for (int e = 0; e < 8; ++e) {
        const float ev = expf((float)c[e] - mx);
        s += ev;
        c[e] = (f16)ev;
      }
      rowv[j] = c;
    }
    const float inv = 1.0f / s;
    for (int j = 0; j < 32; ++j) {
      v8h c = rowv[j];
#pragma unroll
      for (int e = 0; e < 8; ++e) c[e] = (f16)((float)c[e] * inv);
      rowv[j] = c;
    }
  }
  __syncthreads();

  // ---- Phase E: O = (P @ V + pe(q)) * z  -> OF (reuses XN region) ----
  stage_weight_swizzled(WB, Wproj, CDIM, CDIM, tid); // stage for Phase F
  f16* OF = XN;
  for (int t = wave; t < 64; t += 8) {
    const int mt = (t >> 2) << 4, nt = (t & 3) << 4;
    v8f acc = {};
#pragma unroll
    for (int kk = 0; kk < NTOK; kk += 32) {
      v16h a  = frag_a_rowmajor(SM, NTOK, mt, kk, lane);
      v16h bm = frag_b_swizzled(VF, 4, kk, nt, lane);
      acc = __builtin_amdgcn_wmma_f32_16x16x32_f16(false, a, false, bm,
                                                   (short)0, acc, false, false);
    }
    const int chan = nt + (lane & 15);
    const int mb   = mt + ((lane >> 4) << 3);
    const float peb = pe_b[chan];
    float w9[9];
#pragma unroll
    for (int j = 0; j < 9; ++j) w9[j] = pe_w[chan * 9 + j];
#pragma unroll
    for (int r = 0; r < 8; ++r) {
      const int tok = mb + r;
      const int py = tok >> 4, px = tok & 15;
      float pe = peb;
#pragma unroll
      for (int dy = 0; dy < 3; ++dy) {
        const int yy = py + dy - 1;
        if ((unsigned)yy < 16u) {
#pragma unroll
          for (int dx = 0; dx < 3; ++dx) {
            const int xx = px + dx - 1;
            if ((unsigned)xx < 16u)
              pe += (float)QF[((yy << 4) | xx) * CDIM + chan] * w9[dy * 3 + dx];
          }
        }
      }
      const float val = (acc[r] + pe) * (float)ZF[tok * CDIM + chan];
      OF[tok * CDIM + chan] = (f16)val;
    }
  }
  __syncthreads();

  // ---- Phase F: final = O @ Wproj + residual; overlap-add via atomics ----
  for (int t = wave; t < 64; t += 8) {
    const int mt = (t >> 2) << 4, nt = (t & 3) << 4;
    const v8f acc = gemm64_tile(OF, WB, mt, nt, lane);
    const int chan = nt + (lane & 15);
    const int mb   = mt + ((lane >> 4) << 3);
#pragma unroll
    for (int r = 0; r < 8; ++r) {
      const int tok = mb + r;
      const int py = tok >> 4, px = tok & 15;
      const size_t gi = (size_t)(b * CDIM + chan) * HWSZ +
                        (size_t)(top + py) * WDIM + (left + px);
      atomicAdd(&yacc[gi], acc[r] + x[gi]);
    }
  }
}

// ---------------- kernel 2: mask * LN -> h1 = gelu(xn @ fc1 + b) (f16) --------
__global__ __launch_bounds__(256, 1)
void lrsa_ffn1_kernel(const float* __restrict__ yacc, const float* __restrict__ mask,
                      const float* __restrict__ ln2_w, const float* __restrict__ ln2_b,
                      const float* __restrict__ fc1_w, const float* __restrict__ fc1_b,
                      f16* __restrict__ h1) {
  extern __shared__ char smem[];
  f16* XN = (f16*)smem;               // 256x64  = 32KB
  f16* WB = (f16*)(smem + 32768);     // swizzled 64x128 = 16KB
  float* RAWf = (float*)(smem + 49152); // 256 pixels x 64 ch f32 = 64KB
  const int tid = threadIdx.x, lane = tid & 31, wave = tid >> 5;
  const uint32_t ldsbase = __builtin_amdgcn_groupstaticsize();
  const int gbase = blockIdx.x * 256;
  const int b = gbase >> 16;
  const int hwbase = gbase & 65535;

  __builtin_prefetch(fc1_w, 0, 0);

  // async-copy one 256-pixel row segment x 64 channels of yacc into LDS
  {
    const float* yb = yacc + (size_t)b * CDIM * HWSZ + hwbase;
    const uint32_t rawoff = ldsbase + 49152u;
    for (int it = 0; it < 16; ++it) {
      const int f = tid + (it << 8);    // 4096 chunks of 16 bytes
      const int c = f >> 6;
      const int ch = f & 63;
      async_load_b128(rawoff + ((uint32_t)f << 4), yb + (size_t)c * HWSZ + (ch << 2));
    }
    wait_async();
  }
  stage_weight_swizzled(WB, fc1_w, CDIM, C2, tid);
  __syncthreads();

  {
    const int hw = hwbase + tid;
    const int yy = hw >> 8, xx = hw & 255;
    const float msk = mask[yy] * mask[xx];
    float v[CDIM];
    float mu = 0.0f;
#pragma unroll
    for (int c = 0; c < CDIM; ++c) { v[c] = RAWf[(c << 8) + tid] * msk; mu += v[c]; }
    mu *= (1.0f / CDIM);
    float var = 0.0f;
#pragma unroll
    for (int c = 0; c < CDIM; ++c) { const float d = v[c] - mu; var += d * d; }
    var *= (1.0f / CDIM);
    const float rs = rsqrtf(var + 1e-6f);
#pragma unroll
    for (int c = 0; c < CDIM; ++c)
      XN[tid * CDIM + c] = (f16)((v[c] - mu) * rs * ln2_w[c] + ln2_b[c]);
  }
  __syncthreads();

  for (int t = wave; t < 128; t += 8) {
    const int mt = (t >> 3) << 4, nt = (t & 7) << 4;
    v8f acc = {};
#pragma unroll
    for (int kk = 0; kk < CDIM; kk += 32) {
      v16h a  = frag_a_rowmajor(XN, CDIM, mt, kk, lane);
      v16h bm = frag_b_swizzled(WB, 8, kk, nt, lane);
      acc = __builtin_amdgcn_wmma_f32_16x16x32_f16(false, a, false, bm,
                                                   (short)0, acc, false, false);
    }
    const int col = nt + (lane & 15);
    const int mb  = mt + ((lane >> 4) << 3);
    const float bias = fc1_b[col];
#pragma unroll
    for (int r = 0; r < 8; ++r)
      h1[(size_t)(gbase + mb + r) * C2 + col] = (f16)gelu_exact(acc[r] + bias);
  }
}

// ---------------- kernel 3: dw5x5 + gelu + fc2 + residual ---------------------
__global__ __launch_bounds__(256, 1)
void lrsa_ffn2_kernel(const f16* __restrict__ h1, const float* __restrict__ yacc,
                      const float* __restrict__ mask,
                      const float* __restrict__ dw_w, const float* __restrict__ dw_b,
                      const float* __restrict__ fc2_w, const float* __restrict__ fc2_b,
                      float* __restrict__ out) {
  extern __shared__ char smem[];
  f16*   HT = (f16*)smem;                 // 400x128 f16 = 102400
  f16*   SB = (f16*)(smem + 102400);      // 256x128      =  65536
  f16*   WB = (f16*)(smem + 167936);      // swizzled 128x64 = 16384
  float* WD = (float*)(smem + 184320);    // dw weights transposed [25][128] f32
  const int tid = threadIdx.x, lane = tid & 31, wave = tid >> 5;
  const uint32_t ldsbase = __builtin_amdgcn_groupstaticsize();
  const int b = blockIdx.x >> 8;
  const int t = blockIdx.x & 255;
  const int y0 = (t >> 4) << 4, x0 = (t & 15) << 4;

  __builtin_prefetch(fc2_w, 0, 0);
  __builtin_prefetch(dw_w, 0, 0);

  // haloed h1 tile (20x20 x 128ch) via async b128; zero pad at image border
  for (int f = tid; f < HALO * HALO * 16; f += 256) {
    const int e = f >> 4, ch = f & 15;
    const int gy = y0 + e / HALO - 2;
    const int gx = x0 + e % HALO - 2;
    if ((unsigned)gy < (unsigned)HDIM && (unsigned)gx < (unsigned)WDIM) {
      async_load_b128(ldsbase + ((uint32_t)f << 4),
                      h1 + ((size_t)(b * HWSZ + gy * WDIM + gx) << 7) + (ch << 3));
    } else {
      const int4 z = {0, 0, 0, 0};
      *(int4*)(smem + ((size_t)f << 4)) = z;
    }
  }
  stage_weight_swizzled(WB, fc2_w, C2, CDIM, tid);
  for (int i = tid; i < 25 * C2; i += 256) {
    const int tap = i >> 7, c = i & 127;
    WD[i] = dw_w[c * 25 + tap];
  }
  wait_async();
  __syncthreads();

  // s = h1 + gelu(dwconv5x5(h1)), 4 channels per step (b64 data + b128 weights)
  {
    const int p = tid;
    const int py = p >> 4, px = p & 15;
    const f16* base = HT + (py * HALO + px) * C2; // origin already at (-2,-2)
    for (int c4 = 0; c4 < C2; c4 += 4) {
      float4 accv = *(const float4*)(dw_b + c4);
#pragma unroll
      for (int dy = 0; dy < 5; ++dy)
#pragma unroll
        for (int dx = 0; dx < 5; ++dx) {
          const v4h d = *(const v4h*)(base + (dy * HALO + dx) * C2 + c4);
          const float4 w = *(const float4*)(WD + (dy * 5 + dx) * C2 + c4);
          accv.x += (float)d[0] * w.x;
          accv.y += (float)d[1] * w.y;
          accv.z += (float)d[2] * w.z;
          accv.w += (float)d[3] * w.w;
        }
      const v4h ctr = *(const v4h*)(base + (2 * HALO + 2) * C2 + c4);
      v4h o;
      o[0] = (f16)((float)ctr[0] + gelu_exact(accv.x));
      o[1] = (f16)((float)ctr[1] + gelu_exact(accv.y));
      o[2] = (f16)((float)ctr[2] + gelu_exact(accv.z));
      o[3] = (f16)((float)ctr[3] + gelu_exact(accv.w));
      *(v4h*)(SB + p * C2 + c4) = o;
    }
  }
  __syncthreads();

  // out = s @ fc2 + fc2_b + xt
  for (int tt = wave; tt < 64; tt += 8) {
    const int mt = (tt >> 2) << 4, nt = (tt & 3) << 4;
    v8f acc = {};
#pragma unroll
    for (int kk = 0; kk < C2; kk += 32) {
      v16h a  = frag_a_rowmajor(SB, C2, mt, kk, lane);
      v16h bm = frag_b_swizzled(WB, 4, kk, nt, lane);
      acc = __builtin_amdgcn_wmma_f32_16x16x32_f16(false, a, false, bm,
                                                   (short)0, acc, false, false);
    }
    const int chan = nt + (lane & 15);
    const int mb   = mt + ((lane >> 4) << 3);
    const float bias = fc2_b[chan];
#pragma unroll
    for (int r = 0; r < 8; ++r) {
      const int p = mb + r;
      const int gy = y0 + (p >> 4), gx = x0 + (p & 15);
      const size_t gi = (size_t)(b * CDIM + chan) * HWSZ + (size_t)gy * WDIM + gx;
      const float xt = yacc[gi] * mask[gy] * mask[gx];
      out[gi] = acc[r] + bias + xt;
    }
  }
}

extern "C" void kernel_launch(void* const* d_in, const int* in_sizes, int n_in,
                              void* d_out, int out_size, void* d_ws, size_t ws_size,
                              hipStream_t stream) {
  (void)in_sizes; (void)n_in; (void)out_size; (void)ws_size;
  const float* x     = (const float*)d_in[0];
  // d_in[1] = ps (int scalar) — dimensions are compiled in (ps == 16)
  const float* ln1_w = (const float*)d_in[2];
  const float* ln1_b = (const float*)d_in[3];
  const float* Wg    = (const float*)d_in[4];
  const float* bg    = (const float*)d_in[5];
  const float* Wq    = (const float*)d_in[6];
  const float* Wk    = (const float*)d_in[7];
  const float* Wv    = (const float*)d_in[8];
  const float* pe_w  = (const float*)d_in[9];
  const float* pe_b  = (const float*)d_in[10];
  const float* Wproj = (const float*)d_in[11];
  const float* ln2_w = (const float*)d_in[12];
  const float* ln2_b = (const float*)d_in[13];
  const float* fc1_w = (const float*)d_in[14];
  const float* fc1_b = (const float*)d_in[15];
  const float* dw_w  = (const float*)d_in[16];
  const float* dw_b  = (const float*)d_in[17];
  const float* fc2_w = (const float*)d_in[18];
  const float* fc2_b = (const float*)d_in[19];
  float* out = (float*)d_out;

  char* ws = (char*)d_ws;
  float* yacc = (float*)ws;                       // B*C*H*W f32 = 64 MiB
  float* mask = (float*)(ws + 67108864);          // 256 f32
  f16*   h1   = (f16*)(ws + 67112960);            // B*HW*2C f16 = 64 MiB

  lrsa_init_kernel<<<BDIM * CDIM * HWSZ / 256, 256, 0, stream>>>(yacc, mask);
  lrsa_attn_kernel<<<BDIM * NPH * NPW, 256, 303104, stream>>>(
      x, ln1_w, ln1_b, Wg, bg, Wq, Wk, Wv, pe_w, pe_b, Wproj, yacc);
  lrsa_ffn1_kernel<<<BDIM * HWSZ / 256, 256, 114688, stream>>>(
      yacc, mask, ln2_w, ln2_b, fc1_w, fc1_b, h1);
  lrsa_ffn2_kernel<<<BDIM * HWSZ / 256, 256, 197120, stream>>>(
      h1, yacc, mask, dw_w, dw_b, fc2_w, fc2_b, out);
}
